// RelGraphConv_87634512707600
// MI455X (gfx1250) — matrix operand order repaired and checked by reading, not compiled, not run on hardware
//
#include <hip/hip_runtime.h>
#include <hip/hip_bf16.h>

// ---------------------------------------------------------------------------
// RGCN layer for MI455X (gfx1250, wave32, WMMA).
//   Hr[n,r,:] = feat[n,:] @ W_r          (bf16 WMMA, f32 acc, Hr stored bf16)
//   agg[d,:] += norm_e * Hr[src_e, et_e] (hw f32 atomics, agg fits in L2)
//   out = relu(agg + feat @ loop_w + b)  (bf16 WMMA + fused epilogue)
// ---------------------------------------------------------------------------

#define N_NODES   50000
#define N_EDGES   800000
#define IN_FEAT   256
#define OUT_FEAT  256
#define NUM_RELS  8
#define NUM_BASES 4

typedef __bf16          v16bf  __attribute__((ext_vector_type(16)));
typedef float           v8f    __attribute__((ext_vector_type(8)));
typedef unsigned short  u16x4  __attribute__((ext_vector_type(4)));
typedef unsigned short  u16x8  __attribute__((ext_vector_type(8)));
typedef unsigned short  u16x16 __attribute__((ext_vector_type(16)));

__device__ __forceinline__ unsigned short f2bf(float f) {
    unsigned int u = __float_as_uint(f);
    u += 0x7FFFu + ((u >> 16) & 1u);           // round-to-nearest-even
    return (unsigned short)(u >> 16);
}
__device__ __forceinline__ float bf2f(unsigned short h) {
    return __uint_as_float(((unsigned int)h) << 16);
}

// A fragment (16x32 bf16): lane h=lane>>4, row m=lane&15.
// halves 0..7  = K in [kLo,  kLo+8)   (= K0 + h*16)
// halves 8..15 = K in [kHi,  kHi+8)   (= K0 + h*16 + 8)
__device__ __forceinline__ v16bf load_a_frag(const unsigned short* row, int kLo, int kHi) {
    u16x8 lo = *(const u16x8*)(row + kLo);
    u16x8 hi = *(const u16x8*)(row + kHi);
    u16x16 t;
#pragma unroll
    for (int i = 0; i < 8; ++i) { t[i] = lo[i]; t[i + 8] = hi[i]; }
    return __builtin_bit_cast(v16bf, t);
}

// ---------------------------------------------------------------------------
// Kernel 1a: W_r = sum_b w_comp[r,b] * weight[b,:,:], stored TRANSPOSED as
// bf16 Wt[r][o][i] so each WMMA B-fragment is one contiguous 32B load.
// ---------------------------------------------------------------------------
__global__ void __launch_bounds__(256)
k_make_wt(const float* __restrict__ weight, const float* __restrict__ w_comp,
          unsigned short* __restrict__ Wt) {
    int gid = blockIdx.x * 256 + threadIdx.x;          // over R*OUT*IN
    if (gid >= NUM_RELS * OUT_FEAT * IN_FEAT) return;
    int i = gid & (IN_FEAT - 1);
    int o = (gid >> 8) & (OUT_FEAT - 1);
    int r = gid >> 16;
    float acc = 0.f;
#pragma unroll
    for (int b = 0; b < NUM_BASES; ++b)
        acc += w_comp[r * NUM_BASES + b] * weight[(b * IN_FEAT + i) * OUT_FEAT + o];
    Wt[gid] = f2bf(acc);                               // gid == (r*OUT+o)*IN+i
}

// Kernel 1b: loopT[o][i] = bf16(loop_weight[i][o])
__global__ void __launch_bounds__(256)
k_make_loopt(const float* __restrict__ lw, unsigned short* __restrict__ loopT) {
    int gid = blockIdx.x * 256 + threadIdx.x;          // over OUT*IN
    if (gid >= OUT_FEAT * IN_FEAT) return;
    int i = gid & (IN_FEAT - 1);
    int o = gid >> 8;
    loopT[gid] = f2bf(lw[i * OUT_FEAT + o]);
}

// Kernel 1c: feat -> bf16 (row major, 4 elems / thread)
__global__ void __launch_bounds__(256)
k_feat_bf16(const float* __restrict__ feat, unsigned short* __restrict__ fb) {
    long gid = (long)blockIdx.x * 256 + threadIdx.x;   // over N*IN/4
    if (gid >= (long)N_NODES * IN_FEAT / 4) return;
    const float4 v = ((const float4*)feat)[gid];
    u16x4 o; o[0] = f2bf(v.x); o[1] = f2bf(v.y); o[2] = f2bf(v.z); o[3] = f2bf(v.w);
    ((u16x4*)fb)[gid] = o;
}

// ---------------------------------------------------------------------------
// Kernel 2: Hr[n][r][o] = feat[n,:] @ W_r   (bf16 in, f32 acc, bf16 out)
// One block per 16-node strip (3125 strips). Strip staged once in LDS and
// shared by all 8 waves; wave w computes relation r=w, all 16 out-tiles.
// ---------------------------------------------------------------------------
__global__ void __launch_bounds__(256)
k_relgemm(const unsigned short* __restrict__ featB,
          const unsigned short* __restrict__ Wt,
          unsigned short* __restrict__ Hr) {
    __shared__ unsigned short sA[16 * IN_FEAT];        // 8 KB
    const long nodeBase = (long)blockIdx.x * 16;

    {   // cooperative 16x256 bf16 strip load (512 x uint4)
        const uint4* src = (const uint4*)(featB + nodeBase * IN_FEAT);
        uint4* dst = (uint4*)sA;
        for (int idx = threadIdx.x; idx < 16 * IN_FEAT / 8; idx += 256)
            dst[idx] = src[idx];
    }
    __syncthreads();

    const int lane = threadIdx.x & 31;
    const int r    = threadIdx.x >> 5;                 // wave id == relation
    const int h    = lane >> 4;
    const int m    = lane & 15;

    v16bf a[8];
#pragma unroll
    for (int kb = 0; kb < 8; ++kb) {
        const int k0 = kb * 32;
        a[kb] = load_a_frag(sA + m * IN_FEAT, k0 + h * 16, k0 + h * 16 + 8);
    }

    const unsigned short* wbase = Wt + (long)r * OUT_FEAT * IN_FEAT;
    for (int ot = 0; ot < 16; ++ot) {
        const int o = ot * 16 + m;                     // column held by lane
        const unsigned short* col = wbase + (long)o * IN_FEAT + h * 16;
        v8f acc = {};
#pragma unroll
        for (int kb = 0; kb < 8; ++kb) {
            v16bf b = __builtin_bit_cast(v16bf, *(const u16x16*)(col + kb * 32));
            acc = __builtin_amdgcn_wmma_f32_16x16x32_bf16(
                      false, a[kb], false, b, (short)0, acc, false, false);
        }
#pragma unroll
        for (int v = 0; v < 8; ++v) {                  // D: row = v + 8h, col = m
            const long mm = nodeBase + v + 8 * h;
            Hr[(mm * NUM_RELS + r) * OUT_FEAT + ot * 16 + m] = f2bf(acc[v]);
        }
    }
}

// ---------------------------------------------------------------------------
// Kernel 3: per-edge gather + norm scale + scatter-add.
// 4 edges per block-iteration, 64 lanes/edge, u16x4 gathers (8B/lane),
// agg (51.2 MB) is L2-resident -> hw f32 atomics.
// ---------------------------------------------------------------------------
__global__ void __launch_bounds__(256)
k_edges(const unsigned short* __restrict__ Hr,
        const float* __restrict__ norm,
        const int* __restrict__ src,
        const int* __restrict__ dst,
        const int* __restrict__ ety,
        float* __restrict__ agg) {
    const int sub = threadIdx.x >> 6;                  // edge within quad
    const int fg  = threadIdx.x & 63;                  // feature group (x4)
    for (int q = blockIdx.x; q < N_EDGES / 4; q += gridDim.x) {
        const int e = q * 4 + sub;
        const int s = src[e], d = dst[e], r = ety[e];
        const float nv = norm[e];
        const u16x4 hv = *(const u16x4*)(Hr + ((long)s * NUM_RELS + r) * OUT_FEAT + fg * 4);
        float* ap = agg + (long)d * OUT_FEAT + fg * 4;
#pragma unroll
        for (int j = 0; j < 4; ++j)
            unsafeAtomicAdd(ap + j, bf2f(hv[j]) * nv);
    }
}

// ---------------------------------------------------------------------------
// Kernel 4: out = relu(agg + feat @ loop_w + bias). Same WMMA strip scheme;
// 16 out-tiles split 2-per-wave, epilogue fused.
// ---------------------------------------------------------------------------
__global__ void __launch_bounds__(256)
k_selfloop(const unsigned short* __restrict__ featB,
           const unsigned short* __restrict__ loopT,
           const float* __restrict__ agg,
           const float* __restrict__ bias,
           float* __restrict__ out) {
    __shared__ unsigned short sA[16 * IN_FEAT];
    const long nodeBase = (long)blockIdx.x * 16;
    {
        const uint4* src = (const uint4*)(featB + nodeBase * IN_FEAT);
        uint4* dst = (uint4*)sA;
        for (int idx = threadIdx.x; idx < 16 * IN_FEAT / 8; idx += 256)
            dst[idx] = src[idx];
    }
    __syncthreads();

    const int lane = threadIdx.x & 31;
    const int wave = threadIdx.x >> 5;
    const int h    = lane >> 4;
    const int m    = lane & 15;

    v16bf a[8];
#pragma unroll
    for (int kb = 0; kb < 8; ++kb) {
        const int k0 = kb * 32;
        a[kb] = load_a_frag(sA + m * IN_FEAT, k0 + h * 16, k0 + h * 16 + 8);
    }

#pragma unroll
    for (int t = 0; t < 2; ++t) {
        const int ot = wave * 2 + t;
        const int o  = ot * 16 + m;
        const unsigned short* col = loopT + (long)o * IN_FEAT + h * 16;
        v8f acc = {};
#pragma unroll
        for (int kb = 0; kb < 8; ++kb) {
            v16bf b = __builtin_bit_cast(v16bf, *(const u16x16*)(col + kb * 32));
            acc = __builtin_amdgcn_wmma_f32_16x16x32_bf16(
                      false, a[kb], false, b, (short)0, acc, false, false);
        }
        const float bb = bias[o];
#pragma unroll
        for (int v = 0; v < 8; ++v) {
            const long mm = nodeBase + v + 8 * h;
            const long idx = mm * OUT_FEAT + o;
            out[idx] = fmaxf(acc[v] + agg[idx] + bb, 0.f);
        }
    }
}

// ---------------------------------------------------------------------------
extern "C" void kernel_launch(void* const* d_in, const int* in_sizes, int n_in,
                              void* d_out, int out_size, void* d_ws, size_t ws_size,
                              hipStream_t stream) {
    const float* feat   = (const float*)d_in[0];
    const float* weight = (const float*)d_in[1];
    const float* w_comp = (const float*)d_in[2];
    const float* loop_w = (const float*)d_in[3];
    const float* h_bias = (const float*)d_in[4];
    const float* norm   = (const float*)d_in[5];
    const int*   src    = (const int*)d_in[6];
    const int*   dst    = (const int*)d_in[7];
    const int*   ety    = (const int*)d_in[8];
    float* out = (float*)d_out;

    // workspace carve-up (total ~283 MB)
    unsigned short* Wt    = (unsigned short*)d_ws;                       // 1 MB
    unsigned short* loopT = Wt    + (size_t)NUM_RELS * OUT_FEAT * IN_FEAT;   // 128 KB
    unsigned short* featB = loopT + (size_t)OUT_FEAT * IN_FEAT;              // 25.6 MB
    unsigned short* Hr    = featB + (size_t)N_NODES * IN_FEAT;               // 204.8 MB
    float*          agg   = (float*)(Hr + (size_t)N_NODES * NUM_RELS * OUT_FEAT); // 51.2 MB

    hipMemsetAsync(agg, 0, (size_t)N_NODES * OUT_FEAT * sizeof(float), stream);

    k_make_wt   <<<(NUM_RELS * OUT_FEAT * IN_FEAT + 255) / 256, 256, 0, stream>>>(weight, w_comp, Wt);
    k_make_loopt<<<(OUT_FEAT * IN_FEAT + 255) / 256,            256, 0, stream>>>(loop_w, loopT);
    k_feat_bf16 <<<(N_NODES * IN_FEAT / 4 + 255) / 256,         256, 0, stream>>>(feat, featB);

    k_relgemm   <<<N_NODES / 16, 256, 0, stream>>>(featB, Wt, Hr);
    k_edges     <<<8192,          256, 0, stream>>>(Hr, norm, src, dst, ety, agg);
    k_selfloop  <<<N_NODES / 16, 256, 0, stream>>>(featB, loopT, agg, h_bias, out);
}
